// LSTMLayer_60120952209952
// MI455X (gfx1250) — matrix operand order repaired and checked by reading, not compiled, not run on hardware
//
#include <hip/hip_runtime.h>

// Bidirectional LSTM, T=2048, B=32, D=512, H=512.
// Strategy:
//  - Persistent kernel: grid = (32 column-slice WGs) x (2 directions) = 64 WGs,
//    256 threads (8 wave32) each. One device-wide spin barrier per timestep.
//  - Each WG owns 16 h-columns => 64 gate columns (16 per gate i/f/g/o).
//    Its weight slice [64 rows][K=1024] (w_ih || w_hh) is converted to bf16 and
//    parked in LDS once (132KB) -> recurrence never re-reads weights from memory.
//  - Per step: A = [x_t ; h_{t-1}] (32x1024 bf16) staged in LDS; each wave
//    computes one 16x16 f32 tile via 32x v_wmma_f32_16x16x32_bf16 (K=1024).
//  - Gates exchanged via LDS, cell state in registers, h published bf16 to a
//    global buffer in d_ws, grid barrier, next step.
//  - Branch-free sigmoid/tanh (v_exp_f32 + v_rcp_f32) keep the serial
//    elementwise phase off the exec-mask divergence path; x_{t+1} is
//    prefetched (global_prefetch_b8) during the current step.
// ws usage: 256 B barrier area + 64KB h buffer. Requires all 64 WGs resident.

#define T_DIM 2048
#define B_DIM 32
#define D_DIM 512
#define H_DIM 512
#define NWG   64   // total workgroups participating in grid barrier

typedef __attribute__((ext_vector_type(16))) __bf16 v16bf;
typedef __attribute__((ext_vector_type(8)))  float  v8f;

union Frag  { v16bf v; uint4 q[2]; };
union CTile { v8f   v; float f[8]; };

__device__ __forceinline__ unsigned short f2bf(float f) {
    unsigned u = __float_as_uint(f);
    unsigned r = u + 0x7FFFu + ((u >> 16) & 1u);   // round-to-nearest-even
    return (unsigned short)(r >> 16);
}
__device__ __forceinline__ unsigned pack2(float a, float b) {
    return (unsigned)f2bf(a) | ((unsigned)f2bf(b) << 16);
}
// Branch-free activations: v_exp_f32 + v_rcp_f32, no libm range-reduction branches.
__device__ __forceinline__ float fast_sigmoid(float x) {
    return __builtin_amdgcn_rcpf(1.0f + __expf(-x));
}
__device__ __forceinline__ float fast_tanh(float x) {
    return 2.0f * __builtin_amdgcn_rcpf(1.0f + __expf(-2.0f * x)) - 1.0f;
}

__device__ __forceinline__ void grid_barrier(unsigned* bar, unsigned* genLocal) {
    __threadfence();
    __syncthreads();
    if (threadIdx.x == 0) {
        unsigned g = *genLocal;
        volatile unsigned* vcnt = bar;
        volatile unsigned* vgen = bar + 1;
        unsigned old = atomicAdd(bar, 1u);
        if (old == NWG - 1u) {
            *vcnt = 0u;                 // reset for next round (others are spinning on gen)
            __threadfence();
            atomicAdd(bar + 1, 1u);     // release
        } else {
            while (*vgen <= g) { __builtin_amdgcn_s_sleep(2); }
        }
        *genLocal = g + 1u;
    }
    __syncthreads();
    __threadfence();
}

__global__ void init_barrier_kernel(unsigned* bar) {
    if (threadIdx.x < 2) bar[threadIdx.x] = 0u;
}

__global__ __launch_bounds__(256, 1)
void lstm_bidir_persistent(const float* __restrict__ x,
                           const float* __restrict__ h0,
                           const float* __restrict__ c0,
                           const float* __restrict__ wih_f, const float* __restrict__ whh_f,
                           const float* __restrict__ bias_f,
                           const float* __restrict__ wih_b, const float* __restrict__ whh_b,
                           const float* __restrict__ bias_b,
                           float* __restrict__ out,
                           unsigned short* __restrict__ hbuf,   // [2][B][H] bf16 shared h
                           unsigned* __restrict__ bar)
{
    // LDS: padded row stride (1032) breaks the 2048B same-bank stride on fragment loads.
    __shared__ unsigned short sW[64][1032];   // 132KB  weight slice rows: r = gate*16+jj, K = [w_ih(512) | w_hh(512)]
    __shared__ unsigned short sA[32][1032];   // 66KB   A rows: batch,      K = [x_t(512)  | h_{t-1}(512)]
    __shared__ float          sG[4][32][16];  // 8KB    pre-activation gate tiles

    const int tid   = threadIdx.x;
    const int lane  = tid & 31;
    const int wave  = tid >> 5;        // 0..7
    const int mTile = wave >> 2;       // 0..1  (batch halves)
    const int gate  = wave & 3;        // 0..3  (i,f,g,o)
    const int dir   = blockIdx.y;      // 0 fwd, 1 bwd
    const int j0    = blockIdx.x * 16; // owned h-column base

    const float* wih  = dir ? wih_b  : wih_f;
    const float* whh  = dir ? whh_b  : whh_f;
    const float* bias = dir ? bias_b : bias_f;

    // ---- park weight slice in LDS as bf16 (done once) ----
    for (int i = tid; i < 64 * 128; i += 256) {
        int r  = i >> 7;               // 0..63
        int c4 = (i & 127) << 2;       // 0,4,...,508
        int g = r >> 4, jj = r & 15;
        int grow = g * H_DIM + j0 + jj;           // row in (4H, D/H) weight matrices
        const float4 a = *(const float4*)(wih + (size_t)grow * D_DIM + c4);
        const float4 b = *(const float4*)(whh + (size_t)grow * H_DIM + c4);
        *(uint2*)&sW[r][c4]       = make_uint2(pack2(a.x, a.y), pack2(a.z, a.w));
        *(uint2*)&sW[r][512 + c4] = make_uint2(pack2(b.x, b.y), pack2(b.z, b.w));
    }

    // ---- per-thread cell state: thread owns (bRow, j) and (bRow+16, j) ----
    const int j    = tid & 15;
    const int bRow = tid >> 4;         // 0..15
    const int colG = j0 + j;
    float cellA = c0[(size_t)dir * B_DIM * H_DIM + bRow * H_DIM + colG];
    float cellB = c0[(size_t)dir * B_DIM * H_DIM + (bRow + 16) * H_DIM + colG];
    const float bi = bias[0 * H_DIM + colG];
    const float bf = bias[1 * H_DIM + colG];
    const float bg = bias[2 * H_DIM + colG];
    const float bo = bias[3 * H_DIM + colG];
    float hOutA = 0.f, hOutB = 0.f;

    // publish h0 slice (bf16) so every WG can read full h at step 0
    hbuf[dir * B_DIM * H_DIM + bRow * H_DIM + colG] =
        f2bf(h0[(size_t)dir * B_DIM * H_DIM + bRow * H_DIM + colG]);
    hbuf[dir * B_DIM * H_DIM + (bRow + 16) * H_DIM + colG] =
        f2bf(h0[(size_t)dir * B_DIM * H_DIM + (bRow + 16) * H_DIM + colG]);

    unsigned genLocal = 0;
    grid_barrier(bar, &genLocal);

    // precomputed fragment indexing (ISA 16-bit WMMA layouts)
    const int aRow  = mTile * 16 + (lane & 15);
    const int halfA = (lane >> 4) << 3;     // A: lanes 0-15 hold K {0..7,16..23}, lanes 16-31 {8..15,24..31}
    const int bRowW = gate * 16 + (lane & 15);
    const int halfB = (lane >> 4) << 3;     // B: lanes 0-15 hold K {0..7,16..23}, lanes 16-31 {8..15,24..31}

    for (int t = 0; t < T_DIM; ++t) {
        const int tf = dir ? (T_DIM - 1 - t) : t;
        // next step's time index (clamped) for prefetch
        const int tfn = dir ? (tf > 0 ? tf - 1 : 0) : (tf < T_DIM - 1 ? tf + 1 : tf);

        // stage x_t (fp32 -> bf16) into sA[:, 0:512]; prefetch x_{t+1} behind it
        const float* xt = x + (size_t)tf  * (B_DIM * D_DIM);
        const float* xn = x + (size_t)tfn * (B_DIM * D_DIM);
        for (int i = tid; i < 32 * 128; i += 256) {
            int b  = i >> 7;
            int c4 = (i & 127) << 2;
            float4 v = *(const float4*)(xt + b * D_DIM + c4);
            *(uint2*)&sA[b][c4] = make_uint2(pack2(v.x, v.y), pack2(v.z, v.w));
            __builtin_prefetch(xn + b * D_DIM + c4, 0, 0);   // global_prefetch_b8
        }
        // stage h_{t-1} (bf16 copy) into sA[:, 512:1024]
        const unsigned short* hsrc = hbuf + dir * B_DIM * H_DIM;
        for (int i = tid; i < 32 * 64; i += 256) {
            int b  = i >> 6;
            int c8 = (i & 63) << 3;
            *(uint4*)&sA[b][512 + c8] = *(const uint4*)(hsrc + b * H_DIM + c8);
        }
        __syncthreads();

        // ---- 16x16 tile GEMM over K=1024: 32 chained bf16 WMMAs ----
        v8f acc = {};
        #pragma unroll 8
        for (int kt = 0; kt < 32; ++kt) {
            const int kb = kt << 5;
            Frag A, Bm;
            A.q[0]  = *(const uint4*)&sA[aRow][kb + halfA];
            A.q[1]  = *(const uint4*)&sA[aRow][kb + 16 + halfA];
            Bm.q[0] = *(const uint4*)&sW[bRowW][kb + halfB];
            Bm.q[1] = *(const uint4*)&sW[bRowW][kb + 16 + halfB];
            acc = __builtin_amdgcn_wmma_f32_16x16x32_bf16(
                false, A.v, false, Bm.v, (short)0, acc, false, false);
        }

        // scatter C tile to gate exchange buffer (C layout: VGPR r -> M=r(+8), N=lane&15)
        CTile ct; ct.v = acc;
        #pragma unroll
        for (int r = 0; r < 8; ++r) {
            int m = mTile * 16 + r + ((lane >> 4) << 3);
            sG[gate][m][lane & 15] = ct.f[r];
        }
        __syncthreads();

        // ---- elementwise LSTM update (cell state register-resident, branch-free) ----
        {
            float gi = fast_sigmoid(sG[0][bRow][j] + bi);
            float gf = fast_sigmoid(sG[1][bRow][j] + bf);
            float gg = fast_tanh   (sG[2][bRow][j] + bg);
            float go = fast_sigmoid(sG[3][bRow][j] + bo);
            cellA = gf * cellA + gi * gg;
            hOutA = go * fast_tanh(cellA);

            int b2 = bRow + 16;
            float gi2 = fast_sigmoid(sG[0][b2][j] + bi);
            float gf2 = fast_sigmoid(sG[1][b2][j] + bf);
            float gg2 = fast_tanh   (sG[2][b2][j] + bg);
            float go2 = fast_sigmoid(sG[3][b2][j] + bo);
            cellB = gf2 * cellB + gi2 * gg2;
            hOutB = go2 * fast_tanh(cellB);

            // outputs[t][b][dir*H + col] (fp32)
            size_t obase = (size_t)tf * (B_DIM * 2 * H_DIM);
            out[obase + bRow * (2 * H_DIM) + dir * H_DIM + colG] = hOutA;
            out[obase + b2   * (2 * H_DIM) + dir * H_DIM + colG] = hOutB;
            // publish new h (bf16) for all WGs
            hbuf[dir * B_DIM * H_DIM + bRow * H_DIM + colG] = f2bf(hOutA);
            hbuf[dir * B_DIM * H_DIM + b2   * H_DIM + colG] = f2bf(hOutB);
        }

        grid_barrier(bar, &genLocal);
    }

    // ---- final h_T, c_T tails ----
    const size_t OUT_H = (size_t)T_DIM * B_DIM * 2 * H_DIM;
    const size_t OUT_C = OUT_H + (size_t)2 * B_DIM * H_DIM;
    out[OUT_H + (size_t)dir * B_DIM * H_DIM + bRow * H_DIM + colG]        = hOutA;
    out[OUT_H + (size_t)dir * B_DIM * H_DIM + (bRow + 16) * H_DIM + colG] = hOutB;
    out[OUT_C + (size_t)dir * B_DIM * H_DIM + bRow * H_DIM + colG]        = cellA;
    out[OUT_C + (size_t)dir * B_DIM * H_DIM + (bRow + 16) * H_DIM + colG] = cellB;
}

extern "C" void kernel_launch(void* const* d_in, const int* in_sizes, int n_in,
                              void* d_out, int out_size, void* d_ws, size_t ws_size,
                              hipStream_t stream) {
    const float* x     = (const float*)d_in[0];
    const float* h0    = (const float*)d_in[1];
    const float* c0    = (const float*)d_in[2];
    const float* wih_f = (const float*)d_in[3];
    const float* whh_f = (const float*)d_in[4];
    const float* b_f   = (const float*)d_in[5];
    const float* wih_b = (const float*)d_in[6];
    const float* whh_b = (const float*)d_in[7];
    const float* b_b   = (const float*)d_in[8];

    unsigned*       bar  = (unsigned*)d_ws;                       // [cnt, gen]
    unsigned short* hbuf = (unsigned short*)((char*)d_ws + 256);  // 64KB bf16 h state

    init_barrier_kernel<<<1, 64, 0, stream>>>(bar);

    dim3 grid(H_DIM / 16, 2);   // 32 column slices x 2 directions = 64 WGs
    lstm_bidir_persistent<<<grid, 256, 0, stream>>>(
        x, h0, c0, wih_f, whh_f, b_f, wih_b, whh_b, b_b,
        (float*)d_out, hbuf, bar);
}
